// MLP_66589172957641
// MI455X (gfx1250) — matrix operand order repaired and checked by reading, not compiled, not run on hardware
//
#include <hip/hip_runtime.h>
#include <hip/hip_bf16.h>
#include <stdint.h>

// ---------------------------------------------------------------------------
// SupCon + rating head for MI455X (gfx1250, wave32, WMMA).
//
// rating = sigmoid(emb[idx] @ w + b)                       -> d_out[0..4095]
// supcon = SupConLoss(normalize(emb), labels, T=0.07)      -> d_out[4096]
//
// Key algebra (f = L2-normalized rows, adc = f f^T / T):
//  * row max of adc is the diagonal = 1/T (cos-sim <= 1), so the softmax
//    shift is the constant 1/T: denominators are plain sums of
//    exp2(K2*c - K2), K2 = log2(e)/T, emitted as bare v_exp_f32
//    (args are in [-42, 0.2] -> no denormal guard needed).
//  * sum of positive-pair logits = (1/T)*(f_i . G[label_i] - f_i.f_i) with
//    G[c] = sum of same-class rows -> O(N*D), outside the N^2 loop.
//  * positive count = hist[label_i] - 1 -> precomputed histogram.
// N^2 pass per 16x16x64 tile: 4x B128 loads (double-buffered, ping-pong),
// 2x v_wmma_f32_16x16x32_f16, 8x {fma, v_exp, add}. Diagonal exclusion only
// in the wave-uniform tile j0 == i0. WGP-scope global_prefetch 2 tiles ahead.
// ---------------------------------------------------------------------------

typedef __attribute__((ext_vector_type(16))) _Float16 v16h;
typedef __attribute__((ext_vector_type(8)))  float    v8f;

#define NUM_ITEMS   8192
#define LATENT      64
#define BATCH       4096
#define NUM_CLASSES 50
#define INV_T       14.2857142857142857f      // 1 / 0.07
#define K2EXP       20.6099230401f            // INV_T * log2(e)
#define CSPLIT      8
#define NTILES      (NUM_ITEMS / 16)          // 512 column tiles
#define TILES_PER_CHUNK (NTILES / CSPLIT)     // 64

// ---------------------------------------------------------------------------
// Kernel 1: row-wise L2 normalize; emit f16 matrix (WMMA operands) and f32
// matrix (class sums / per-row dot products). One thread per row.
// ---------------------------------------------------------------------------
__global__ void normalize_kernel(const float* __restrict__ emb,
                                 _Float16* __restrict__ featH,
                                 float* __restrict__ featF) {
    int row = blockIdx.x * blockDim.x + threadIdx.x;
    if (row >= NUM_ITEMS) return;
    const float4* e = (const float4*)(emb + row * LATENT);
    float4 v[16];
    float ss = 0.f;
#pragma unroll
    for (int k = 0; k < 16; ++k) {
        v[k] = e[k];
        ss += v[k].x * v[k].x + v[k].y * v[k].y + v[k].z * v[k].z + v[k].w * v[k].w;
    }
    float inv = 1.0f / fmaxf(sqrtf(ss), 1e-12f);   // matches max(norm, 1e-12)
    float4* of = (float4*)(featF + row * LATENT);
    union H8 { _Float16 h[8]; uint4 u; };
    uint4* oh = (uint4*)(featH + row * LATENT);
#pragma unroll
    for (int k = 0; k < 8; ++k) {
        float4 x = v[2 * k], y = v[2 * k + 1];
        x.x *= inv; x.y *= inv; x.z *= inv; x.w *= inv;
        y.x *= inv; y.y *= inv; y.z *= inv; y.w *= inv;
        of[2 * k] = x; of[2 * k + 1] = y;
        H8 p;
        p.h[0] = (_Float16)x.x; p.h[1] = (_Float16)x.y;
        p.h[2] = (_Float16)x.z; p.h[3] = (_Float16)x.w;
        p.h[4] = (_Float16)y.x; p.h[5] = (_Float16)y.y;
        p.h[6] = (_Float16)y.z; p.h[7] = (_Float16)y.w;
        oh[k] = p.u;
    }
}

// ---------------------------------------------------------------------------
// Kernel 2: per-class feature sums G[c][d] and label histogram.
// ---------------------------------------------------------------------------
__global__ void class_sum_kernel(const float* __restrict__ featF,
                                 const int* __restrict__ labels,
                                 float* __restrict__ G,
                                 float* __restrict__ hist) {
    const int c = blockIdx.x;       // 0..NUM_CLASSES-1
    const int d = threadIdx.x;      // 0..LATENT-1
    float acc = 0.f;
    int   cnt = 0;
    for (int row = 0; row < NUM_ITEMS; ++row) {
        if (labels[row] == c) {
            acc += featF[row * LATENT + d];
            ++cnt;
        }
    }
    G[c * LATENT + d] = acc;
    if (d == 0) hist[c] = (float)cnt;
}

// ---------------------------------------------------------------------------
// Kernel 3: rating head. One thread per batch item.
// ---------------------------------------------------------------------------
__global__ void rating_kernel(const int* __restrict__ idx,
                              const float* __restrict__ emb,
                              const float* __restrict__ w,
                              const float* __restrict__ b,
                              float* __restrict__ out) {
    int t = blockIdx.x * blockDim.x + threadIdx.x;
    if (t >= BATCH) return;
    int it = idx[t];
    const float4* e  = (const float4*)(emb + it * LATENT);
    const float4* wv = (const float4*)w;
    float acc = b[0];
#pragma unroll
    for (int k = 0; k < 16; ++k) {
        float4 a = e[k], c = wv[k];
        acc = fmaf(a.x, c.x, acc);
        acc = fmaf(a.y, c.y, acc);
        acc = fmaf(a.z, c.z, acc);
        acc = fmaf(a.w, c.w, acc);
    }
    out[t] = 1.0f / (1.0f + __expf(-acc));
}

// ---------------------------------------------------------------------------
// Kernel 4: N^2 exp-sum pass via WMMA. grid = (512 row-blocks, 8 col-chunks),
// block = 32 (one wave). C fragment layout (16x16 f32): lane holds column
// N = lane%16; VGPR r holds row M = 8*(lane/16) + r.
// ---------------------------------------------------------------------------
__global__ __launch_bounds__(32)
void supcon_gemm_kernel(const _Float16* __restrict__ featH,
                        float* __restrict__ stS) {
    const int rowBlk = blockIdx.x;          // 0..511
    const int chunk  = blockIdx.y;          // 0..7
    const int lane   = threadIdx.x;         // wave32
    const int half   = lane >> 4;           // 0 | 1
    const int li     = lane & 15;
    const int i0     = rowBlk * 16;

    union F { v16h h; uint4 u[2]; };

    // A fragments for this wave's 16 rows (K=64 -> two 16x32 fragments).
    // A 16x32 as loaded here (contiguous k, same order as B): lane<16: K {0..15} in V0-7;
    // lane>=16: K {16..31}. All loads 16B-aligned (row stride 128B).
    const uint4* ap = (const uint4*)(featH + (i0 + li) * LATENT);
    F a0, a1;
    a0.u[0] = ap[2 * half];
    a0.u[1] = ap[2 * half + 1];
    a1.u[0] = ap[4 + 2 * half];
    a1.u[1] = ap[4 + 2 * half + 1];

    float s[8];
#pragma unroll
    for (int r = 0; r < 8; ++r) s[r] = 0.f;

    const int tile0 = chunk * TILES_PER_CHUNK;
    // This lane's column pointer for tile index t: colbase + t*16 rows.
    const _Float16* colbase = featH + (size_t)(tile0 * 16 + li) * LATENT;

    // B fragments (32x16 f16): lane<16 -> K 0..15 across V0-7, lane>=16 ->
    // K 16..31. Column col's fragment = feat row col (B = F^T).
    auto loadB = [&](int t, F& x0, F& x1) {
        const uint4* bp = (const uint4*)(colbase + (size_t)t * 16 * LATENT);
        x0.u[0] = bp[2 * half];
        x0.u[1] = bp[2 * half + 1];
        x1.u[0] = bp[4 + 2 * half];
        x1.u[1] = bp[4 + 2 * half + 1];
    };
    auto compute = [&](int t, const F& x0, const F& x1) {
        v8f c = {};
        c = __builtin_amdgcn_wmma_f32_16x16x32_f16(false, a0.h, false, x0.h,
                                                   (short)0, c, false, false);
        c = __builtin_amdgcn_wmma_f32_16x16x32_f16(false, a1.h, false, x1.h,
                                                   (short)0, c, false, false);
        const int j0 = (tile0 + t) * 16;
        if (j0 != i0) {                 // steady state: 8x {fma, v_exp, add}
#pragma unroll
            for (int r = 0; r < 8; ++r)
                s[r] += __builtin_amdgcn_exp2f(fmaf(c[r], K2EXP, -K2EXP));
        } else {                        // diagonal tile (wave-uniform branch)
#pragma unroll
            for (int r = 0; r < 8; ++r) {
                float e = __builtin_amdgcn_exp2f(fmaf(c[r], K2EXP, -K2EXP));
                s[r] += (li == 8 * half + r) ? 0.f : e;   // logits_mask
            }
        }
    };

    // Software pipeline: ping-pong B buffers, loads overlap WMMA + epilogue.
    F bA0, bA1, bB0, bB1;
    loadB(0, bA0, bA1);
    for (int t = 0; t < TILES_PER_CHUNK; t += 2) {
        loadB(t + 1, bB0, bB1);         // t+1 <= 63 always (even trip count)
        if (t + 2 < TILES_PER_CHUNK) {  // WGP-scope prefetch, 2 tiles ahead
            const _Float16* pf = colbase + (size_t)(t + 2) * 16 * LATENT;
            asm volatile("global_prefetch_b8 %0, off"
                         :: "v"((unsigned long long)(uintptr_t)pf));
        }
        compute(t, bA0, bA1);
        if (t + 2 < TILES_PER_CHUNK) loadB(t + 2, bA0, bA1);
        compute(t + 1, bB0, bB1);
    }

    // Plain sum across the 16 lanes of each half (xor<16 stays in-half).
#pragma unroll
    for (int r = 0; r < 8; ++r)
        for (int off = 1; off < 16; off <<= 1)
            s[r] += __shfl_xor(s[r], off, 32);

    if (li == 0) {                     // lanes 0 and 16 own 8 rows each
        const int rbase = i0 + 8 * half;
#pragma unroll
        for (int r = 0; r < 8; ++r)
            stS[chunk * NUM_ITEMS + rbase + r] = s[r];
    }
}

// ---------------------------------------------------------------------------
// Kernel 5: per-row loss. s = sum of chunk exp-sums; positives via class-sum
// trick; count via histogram. Deterministic block reduce to 32 partials.
// ---------------------------------------------------------------------------
__global__ void supcon_merge_kernel(const float* __restrict__ stS,
                                    const float* __restrict__ featF,
                                    const int* __restrict__ labels,
                                    const float* __restrict__ G,
                                    const float* __restrict__ hist,
                                    float* __restrict__ partials) {
    const int row = blockIdx.x * blockDim.x + threadIdx.x;   // 8192 threads
    float s = 0.f;
#pragma unroll
    for (int k = 0; k < CSPLIT; ++k) s += stS[k * NUM_ITEMS + row];

    const int lab = labels[row];
    const float4* f = (const float4*)(featF + row * LATENT);
    const float4* g = (const float4*)(G + lab * LATENT);
    float dotG = 0.f, selfd = 0.f;
#pragma unroll
    for (int k = 0; k < 16; ++k) {
        float4 a = f[k], q = g[k];
        dotG  = fmaf(a.x, q.x, fmaf(a.y, q.y, fmaf(a.z, q.z, fmaf(a.w, q.w, dotG))));
        selfd = fmaf(a.x, a.x, fmaf(a.y, a.y, fmaf(a.z, a.z, fmaf(a.w, a.w, selfd))));
    }
    const float cnt = hist[lab] - 1.f;                 // positives excl. self
    const float Ls  = INV_T * (dotG - selfd);          // sum_pos adc
    // logits shift c = 1/T (== reference row max, the diagonal):
    // mlpp = (Ls - cnt/T - cnt*log(s + 1e-6)) / (cnt + 1e-6)
    const float mlpp = (Ls - cnt * INV_T - cnt * __logf(s + 1e-6f)) / (cnt + 1e-6f);
    const float rowloss = -mlpp;                       // T/base_T == 1

    __shared__ float sm[256];
    sm[threadIdx.x] = rowloss;
    __syncthreads();
    for (int st = 128; st > 0; st >>= 1) {
        if (threadIdx.x < st) sm[threadIdx.x] += sm[threadIdx.x + st];
        __syncthreads();
    }
    if (threadIdx.x == 0) partials[blockIdx.x] = sm[0];
}

__global__ void supcon_final_kernel(const float* __restrict__ partials,
                                    float* __restrict__ out) {
    float v = partials[threadIdx.x];        // 32 partials, one wave
    for (int off = 16; off > 0; off >>= 1) v += __shfl_xor(v, off, 32);
    if (threadIdx.x == 0) out[BATCH] = v * (1.0f / NUM_ITEMS);
}

// ---------------------------------------------------------------------------
extern "C" void kernel_launch(void* const* d_in, const int* in_sizes, int n_in,
                              void* d_out, int out_size, void* d_ws, size_t ws_size,
                              hipStream_t stream) {
    (void)in_sizes; (void)n_in; (void)out_size; (void)ws_size;
    const int*   item_indices = (const int*)  d_in[0];
    const int*   labels       = (const int*)  d_in[1];
    const float* emb          = (const float*)d_in[2];
    const float* affine_w     = (const float*)d_in[3];
    const float* affine_b     = (const float*)d_in[4];
    float*       out          = (float*)d_out;

    // Workspace (~3.3 MB):
    //   featH : 8192*64 f16  (1 MB)   WMMA operands
    //   featF : 8192*64 f32  (2 MB)   class sums / dot products
    //   stS   : 8*8192  f32  (256 KB) per-chunk exp-sums
    //   G     : 50*64   f32           class-sum vectors
    //   hist  : 50      f32           label histogram (padded to 64)
    //   part  : 32      f32           block partials
    _Float16* featH = (_Float16*)d_ws;
    float* featF = (float*)((char*)d_ws + (size_t)NUM_ITEMS * LATENT * sizeof(_Float16));
    float* stS   = featF + (size_t)NUM_ITEMS * LATENT;
    float* G     = stS + (size_t)CSPLIT * NUM_ITEMS;
    float* hist  = G + (size_t)NUM_CLASSES * LATENT;
    float* partials = hist + 64;

    normalize_kernel<<<NUM_ITEMS / 256, 256, 0, stream>>>(emb, featH, featF);
    class_sum_kernel<<<NUM_CLASSES, LATENT, 0, stream>>>(featF, labels, G, hist);
    rating_kernel<<<BATCH / 256, 256, 0, stream>>>(item_indices, emb, affine_w,
                                                   affine_b, out);
    dim3 grid(NUM_ITEMS / 16, CSPLIT);
    supcon_gemm_kernel<<<grid, 32, 0, stream>>>(featH, stS);
    supcon_merge_kernel<<<NUM_ITEMS / 256, 256, 0, stream>>>(stS, featF, labels,
                                                             G, hist, partials);
    supcon_final_kernel<<<1, 32, 0, stream>>>(partials, out);
}
